// MSA_1305670058666
// MI455X (gfx1250) — matrix operand order repaired and checked by reading, not compiled, not run on hardware
//
#include <hip/hip_runtime.h>
#include <hip/hip_bf16.h>

// ---------------------------------------------------------------------------
// MI455X (gfx1250) implementation of the Mamba-2D block.
//   - All GEMMs use v_wmma_f32_16x16x32_bf16 with double-buffered LDS tiles:
//     global_load_b128 (next tile) overlaps WMMA (current tile), one barrier
//     per K-step.
//   - fp32->bf16 conversion via __builtin_convertvector -> v_cvt_pk_bf16_f32.
//   - Sequential SSM scan: one lane per (batch, channel) = 16384 lanes.
// ---------------------------------------------------------------------------

typedef __attribute__((ext_vector_type(16))) __bf16 v16bf;
typedef __attribute__((ext_vector_type(8)))  __bf16 bf16x8;
typedef __attribute__((ext_vector_type(8)))  float  f32x8;
typedef __attribute__((ext_vector_type(8)))  float  v8f;

#define BATCH   16
#define LSEQ    256
#define DMODEL  1024
#define DINNER  1024
#define BLROWS  (BATCH * LSEQ)   // 4096
#define NDBL    68               // dtr + 2*n (valid columns)
#define LDDBL   80               // padded leading dim (32B-aligned rows)
#define DTR     64

__device__ __forceinline__ float silu_f(float v) { return v / (1.0f + __expf(-v)); }

__device__ __forceinline__ v16bf frag_join(uint4 a, uint4 b) {
    union { v16bf v; uint4 q[2]; } u;
    u.q[0] = a; u.q[1] = b;
    return u.v;
}

// ---------------------------------------------------------------------------
// WMMA GEMM:  C[M,N] = act( A[M,K] @ Bmat (+ bias) )
//   BT=true : Bmat is weights [N,K] (row-major), i.e. C = A @ W^T
//   BT=false: Bmat is [K,N] row-major,            i.e. C = A @ B
// Block tile 128x64, K-step 32, 8 waves, each wave -> 16 rows x 64 cols.
// Double-buffered LDS (24KB). Requirements (all call sites satisfy them):
// M % 128 == 0, K % 32 == 0, lda/ldb/ldc % 8 == 0 except ldc may be %4.
// ---------------------------------------------------------------------------
template <bool BT, int ACT, bool HAS_BIAS>
__global__ __launch_bounds__(256) void wmma_gemm_kernel(
    const float* __restrict__ A, const float* __restrict__ Bmat,
    const float* __restrict__ bias, float* __restrict__ C,
    int M, int N, int K, int lda, int ldb, int ldc,
    long sA, long sB, long sC)
{
    __shared__ __bf16 As[2][128][32];   // [buf][m][k]
    __shared__ __bf16 Bs[2][64][32];    // [buf][n][k]

    const int t    = threadIdx.x;
    const int wv   = t >> 5;         // wave id 0..7
    const int lane = t & 31;
    const int hf   = lane >> 4;      // lane half (0/1)
    const int lw   = lane & 15;
    const int mbase = blockIdx.y * 128;
    const int nbase = blockIdx.x * 64;

    A    += (long)blockIdx.z * sA;
    Bmat += (long)blockIdx.z * sB;
    C    += (long)blockIdx.z * sC;

    v8f acc[4];
#pragma unroll
    for (int i = 0; i < 4; ++i)
        acc[i] = v8f{0.f, 0.f, 0.f, 0.f, 0.f, 0.f, 0.f, 0.f};

    // A staging: thread covers row (t>>1), 16 contiguous k at ((t&1)*16)
    const int amm = t >> 1;
    const int akb = (t & 1) * 16;
    const float* arow = A + (long)(mbase + amm) * lda + akb;

    // B staging (BT): thread covers n-row (t>>2), 8 contiguous k at ((t&3)*8)
    // B staging (NN): thread covers n-col (t&63), 8 contiguous k at ((t>>6)*8)
    const int bnn = BT ? (t >> 2) : (t & 63);
    const int bkb = BT ? ((t & 3) * 8) : ((t >> 6) * 8);
    const bool bok = (nbase + bnn) < N;

    // stage one K-slab into LDS buffer p
    auto stage = [&](int k0, int p) {
        // A tile: 2x f32x8 load -> convertvector -> 2x ds_store_b128
        const f32x8* asrc = (const f32x8*)(arow + k0);
        if (k0 + 32 < K) __builtin_prefetch((const float*)asrc + 32, 0, 1);
        bf16x8 a0 = __builtin_convertvector(asrc[0], bf16x8);
        bf16x8 a1 = __builtin_convertvector(asrc[1], bf16x8);
        *(bf16x8*)&As[p][amm][akb]     = a0;
        *(bf16x8*)&As[p][amm][akb + 8] = a1;
        // B tile as Bs[n][k]: one ds_store_b128 per thread
        bf16x8 bp;
        if (bok) {
            if (BT) {
                const f32x8* bsrc =
                    (const f32x8*)(Bmat + (long)(nbase + bnn) * ldb + k0 + bkb);
                bp = __builtin_convertvector(bsrc[0], bf16x8);
            } else {
                const float* bsrc = Bmat + (long)(k0 + bkb) * ldb + nbase + bnn;
                f32x8 f;
#pragma unroll
                for (int j = 0; j < 8; ++j) f[j] = bsrc[(long)j * ldb];
                bp = __builtin_convertvector(f, bf16x8);
            }
        } else {
            bp = bf16x8{};
        }
        *(bf16x8*)&Bs[p][bnn][bkb] = bp;
    };

    // compute one K-slab from LDS buffer p
    const int mrow = wv * 16 + lw;
    const int ko   = hf * 8;
    auto compute = [&](int p) {
        // A fragment (ds_load_b128 x2), then ALL B fragments, then 4 WMMAs
        const uint4 aq0 = *(const uint4*)&As[p][mrow][ko];
        const uint4 aq1 = *(const uint4*)&As[p][mrow][16 + ko];
        uint4 bq[4][2];
#pragma unroll
        for (int nt = 0; nt < 4; ++nt) {
            const int nn = nt * 16 + lw;
            bq[nt][0] = *(const uint4*)&Bs[p][nn][ko];
            bq[nt][1] = *(const uint4*)&Bs[p][nn][16 + ko];
        }
        const v16bf av = frag_join(aq0, aq1);
#pragma unroll
        for (int nt = 0; nt < 4; ++nt) {
            acc[nt] = __builtin_amdgcn_wmma_f32_16x16x32_bf16(
                false, av, false, frag_join(bq[nt][0], bq[nt][1]),
                (short)0, acc[nt], false, false);
        }
    };

    // ---- double-buffered main loop: one barrier per K-step ----
    stage(0, 0);
    __syncthreads();
    int p = 0;
    for (int k0 = 32; k0 < K; k0 += 32) {
        stage(k0, p ^ 1);   // overlaps with compute(p)
        compute(p);
        __syncthreads();
        p ^= 1;
    }
    compute(p);

    // ---- epilogue: C layout VGPR r -> M = hf*8 + r, N = lane&15 ----
    // (M always a multiple of 128 at every call site; only guard N)
#pragma unroll
    for (int nt = 0; nt < 4; ++nt) {
        const int col = nbase + nt * 16 + lw;
        if (col >= N) continue;
        const float bv = HAS_BIAS ? bias[col] : 0.0f;
        float* crow = C + (long)(mbase + wv * 16 + hf * 8) * ldc + col;
#pragma unroll
        for (int r = 0; r < 8; ++r) {
            float v = acc[nt][r] + bv;
            if (ACT == 1) v = silu_f(v);
            crow[(long)r * ldc] = v;
        }
    }
}

// ---------------------------------------------------------------------------
// Depthwise causal conv (k=4) over L + silu. xi lives in XZ[:, 0:1024], ld 2048.
// ---------------------------------------------------------------------------
__global__ __launch_bounds__(256) void conv_silu_kernel(
    const float* __restrict__ XZ, const float* __restrict__ convw,
    const float* __restrict__ convb, float* __restrict__ XIC)
{
    const int gid = blockIdx.x * 256 + threadIdx.x;   // over BLROWS*DINNER
    if (gid >= BLROWS * DINNER) return;
    const int d  = gid & (DINNER - 1);
    const int bl = gid >> 10;
    const int l  = bl & (LSEQ - 1);
    const int b  = bl >> 8;
    float acc = convb[d];
#pragma unroll
    for (int j = 0; j < 4; ++j) {
        const int ls = l - 3 + j;
        const float xv = (ls >= 0) ? XZ[((long)(b * LSEQ + ls)) * 2048 + d] : 0.0f;
        acc += xv * convw[d * 4 + j];
    }
    XIC[gid] = silu_f(acc);
}

// ---------------------------------------------------------------------------
// Fused SSM scan: softplus(dt), h = exp(dt*A)*h + dt*B*x, y = h.C + D*x, *silu(z)
// One lane per (b, d): 16384 lanes, fully register-resident recurrence.
// ---------------------------------------------------------------------------
__global__ __launch_bounds__(256) void scan_kernel(
    const float* __restrict__ DT, const float* __restrict__ DBL,
    const float* __restrict__ XIC, const float* __restrict__ XZ,
    const float* __restrict__ bdt, const float* __restrict__ Alog,
    const float* __restrict__ Dp, float* __restrict__ Y)
{
    const int gid = blockIdx.x * 256 + threadIdx.x;   // over BATCH*DINNER
    if (gid >= BATCH * DINNER) return;
    const int d = gid & (DINNER - 1);
    const int b = gid >> 10;

    const float A0 = -__expf(Alog[d * 2 + 0]);
    const float A1 = -__expf(Alog[d * 2 + 1]);
    const float bd = bdt[d];
    const float Dv = Dp[d];
    float h0 = 0.0f, h1 = 0.0f;

    for (int l = 0; l < LSEQ; ++l) {
        const long idx = (long)(b * LSEQ + l);
        float dtv = DT[idx * DINNER + d] + bd;
        dtv = (dtv > 20.0f) ? dtv : __logf(1.0f + __expf(dtv));   // softplus
        const float xv = XIC[idx * DINNER + d];
        const float zv = XZ[idx * 2048 + DINNER + d];
        const float4 bc = *(const float4*)(DBL + idx * LDDBL + DTR); // B0 B1 C0 C1
        h0 = __expf(dtv * A0) * h0 + dtv * bc.x * xv;
        h1 = __expf(dtv * A1) * h1 + dtv * bc.y * xv;
        float y = h0 * bc.z + h1 * bc.w + Dv * xv;
        y *= silu_f(zv);
        Y[idx * DINNER + d] = y;
    }
}

// ---------------------------------------------------------------------------
// LayerNorm over dm=1024, one 256-thread block per row.
// ---------------------------------------------------------------------------
__global__ __launch_bounds__(256) void ln_kernel(
    const float* __restrict__ X, const float* __restrict__ g,
    const float* __restrict__ bta, float* __restrict__ Out)
{
    __shared__ float red[256];
    const int row = blockIdx.x;
    const int t = threadIdx.x;
    const float* x = X + (long)row * DMODEL;

    float s = 0.0f;
    for (int i = t; i < DMODEL; i += 256) s += x[i];
    red[t] = s; __syncthreads();
    for (int o = 128; o > 0; o >>= 1) { if (t < o) red[t] += red[t + o]; __syncthreads(); }
    const float mean = red[0] * (1.0f / DMODEL);
    __syncthreads();

    float v = 0.0f;
    for (int i = t; i < DMODEL; i += 256) { const float dlt = x[i] - mean; v += dlt * dlt; }
    red[t] = v; __syncthreads();
    for (int o = 128; o > 0; o >>= 1) { if (t < o) red[t] += red[t + o]; __syncthreads(); }
    const float inv = rsqrtf(red[0] * (1.0f / DMODEL) + 1e-5f);

    for (int i = t; i < DMODEL; i += 256)
        Out[(long)row * DMODEL + i] = (x[i] - mean) * inv * g[i] + bta[i];
}

__global__ __launch_bounds__(256) void add_silu_kernel(
    const float* __restrict__ Sa, const float* __restrict__ Rb,
    float* __restrict__ Out, int n)
{
    const int gid = blockIdx.x * 256 + threadIdx.x;
    if (gid >= n) return;
    const float v = Sa[gid] + Rb[gid];
    Out[gid] = silu_f(v);
}

// ---------------------------------------------------------------------------
// Host-side: one Mamba pass = 4 WMMA GEMMs + conv + scan
// ---------------------------------------------------------------------------
static void run_mamba(const float* in, const float* Win, const float* convw,
                      const float* convb, const float* Wx, const float* Wdt,
                      const float* bdt, const float* Alog, const float* Dp,
                      const float* Wout, float* XZ, float* XIC, float* DBLb,
                      float* DTb, float* Yb, float* outS, hipStream_t stream)
{
    dim3 blk(256);
    // xz = in @ Win^T : [4096,2048]
    wmma_gemm_kernel<true, 0, false><<<dim3(32, 32, 1), blk, 0, stream>>>(
        in, Win, nullptr, XZ, BLROWS, 2 * DINNER, DMODEL, DMODEL, DMODEL, 2 * DINNER, 0, 0, 0);
    // depthwise conv + silu
    conv_silu_kernel<<<dim3((BLROWS * DINNER) / 256), blk, 0, stream>>>(XZ, convw, convb, XIC);
    // dbl = xi_c @ Wx^T : [4096,68] stored with padded ld=80
    wmma_gemm_kernel<true, 0, false><<<dim3(2, 32, 1), blk, 0, stream>>>(
        XIC, Wx, nullptr, DBLb, BLROWS, NDBL, DINNER, DINNER, DINNER, LDDBL, 0, 0, 0);
    // dt_raw = dbl[:, :64] @ Wdt^T : [4096,1024]
    wmma_gemm_kernel<true, 0, false><<<dim3(16, 32, 1), blk, 0, stream>>>(
        DBLb, Wdt, nullptr, DTb, BLROWS, DINNER, DTR, LDDBL, DTR, DINNER, 0, 0, 0);
    // sequential SSM scan (fused softplus/gates)
    scan_kernel<<<dim3((BATCH * DINNER) / 256), blk, 0, stream>>>(
        DTb, DBLb, XIC, XZ, bdt, Alog, Dp, Yb);
    // out = y @ Wout^T : [4096,1024]
    wmma_gemm_kernel<true, 0, false><<<dim3(16, 32, 1), blk, 0, stream>>>(
        Yb, Wout, nullptr, outS, BLROWS, DMODEL, DINNER, DINNER, DINNER, DMODEL, 0, 0, 0);
}

extern "C" void kernel_launch(void* const* d_in, const int* in_sizes, int n_in,
                              void* d_out, int out_size, void* d_ws, size_t ws_size,
                              hipStream_t stream)
{
    (void)in_sizes; (void)n_in; (void)out_size; (void)ws_size;
    const float* x        = (const float*)d_in[0];
    const float* m1_Win   = (const float*)d_in[1];
    const float* m1_convw = (const float*)d_in[2];
    const float* m1_convb = (const float*)d_in[3];
    const float* m1_Wx    = (const float*)d_in[4];
    const float* m1_Wdt   = (const float*)d_in[5];
    const float* m1_bdt   = (const float*)d_in[6];
    const float* m1_Alog  = (const float*)d_in[7];
    const float* m1_D     = (const float*)d_in[8];
    const float* m1_Wout  = (const float*)d_in[9];
    const float* m2_Win   = (const float*)d_in[10];
    const float* m2_convw = (const float*)d_in[11];
    const float* m2_convb = (const float*)d_in[12];
    const float* m2_Wx    = (const float*)d_in[13];
    const float* m2_Wdt   = (const float*)d_in[14];
    const float* m2_bdt   = (const float*)d_in[15];
    const float* m2_Alog  = (const float*)d_in[16];
    const float* m2_D     = (const float*)d_in[17];
    const float* m2_Wout  = (const float*)d_in[18];
    const float* ln1_g    = (const float*)d_in[19];
    const float* ln1_b    = (const float*)d_in[20];
    const float* ln2_g    = (const float*)d_in[21];
    const float* ln2_b    = (const float*)d_in[22];
    const float* W_mid    = (const float*)d_in[23];
    const float* W_spa    = (const float*)d_in[24];
    const float* b_spa    = (const float*)d_in[25];
    const float* W_res    = (const float*)d_in[26];
    float* out = (float*)d_out;

    // workspace carve-up (fp32 elements)
    float* w   = (float*)d_ws;
    float* XZ  = w;                           // 4096 x 2048
    float* XIC = XZ  + (long)BLROWS * 2048;   // 4096 x 1024
    float* DBLb= XIC + (long)BLROWS * DINNER; // 4096 x 80 (68 valid)
    float* DTb = DBLb+ (long)BLROWS * LDDBL;  // 4096 x 1024
    float* Yb  = DTb + (long)BLROWS * DINNER; // 4096 x 1024
    float* Sb  = Yb  + (long)BLROWS * DINNER; // 4096 x 1024
    float* Rb  = Sb  + (long)BLROWS * DINNER; // 4096 x 1024
    float* R2b = Rb  + (long)BLROWS * DINNER; // 4096 x 1024

    dim3 blk(256);
    const long bstride = (long)LSEQ * DMODEL;  // 262144, per-batch stride

    // ---- Mamba block 1: x -> Sb ----
    run_mamba(x, m1_Win, m1_convw, m1_convb, m1_Wx, m1_Wdt, m1_bdt, m1_Alog,
              m1_D, m1_Wout, XZ, XIC, DBLb, DTb, Yb, Sb, stream);
    // ln1: Sb -> Yb
    ln_kernel<<<dim3(BLROWS), blk, 0, stream>>>(Sb, ln1_g, ln1_b, Yb);
    // channel mix: per batch, Sb = silu(W_mid @ Yb_b)  [256,1024]
    wmma_gemm_kernel<false, 1, false><<<dim3(16, 2, BATCH), blk, 0, stream>>>(
        W_mid, Yb, nullptr, Sb, LSEQ, DMODEL, LSEQ, LSEQ, DMODEL, DMODEL,
        0, bstride, bstride);

    // ---- Mamba block 2: Sb -> Sb ----
    run_mamba(Sb, m2_Win, m2_convw, m2_convb, m2_Wx, m2_Wdt, m2_bdt, m2_Alog,
              m2_D, m2_Wout, XZ, XIC, DBLb, DTb, Yb, Sb, stream);
    // ln2: Sb -> Yb
    ln_kernel<<<dim3(BLROWS), blk, 0, stream>>>(Sb, ln2_g, ln2_b, Yb);

    // ---- Residual branch ----
    // r = silu(x @ W_spa^T + b_spa)
    wmma_gemm_kernel<true, 1, true><<<dim3(16, 32, 1), blk, 0, stream>>>(
        x, W_spa, b_spa, Rb, BLROWS, DMODEL, DMODEL, DMODEL, DMODEL, DMODEL, 0, 0, 0);
    // r = silu(W_res @ r_b) per batch
    wmma_gemm_kernel<false, 1, false><<<dim3(16, 2, BATCH), blk, 0, stream>>>(
        W_res, Rb, nullptr, R2b, LSEQ, DMODEL, LSEQ, LSEQ, DMODEL, DMODEL,
        0, bstride, bstride);

    // ---- Final: out = silu(s + r) ----
    add_silu_kernel<<<dim3((BLROWS * DMODEL) / 256), blk, 0, stream>>>(
        Yb, R2b, out, BLROWS * DMODEL);
}
